// TinyViTBlock_15985868275954
// MI455X (gfx1250) — matrix operand order repaired and checked by reading, not compiled, not run on hardware
//
#include <hip/hip_runtime.h>
#include <hip/hip_bf16.h>
#include <math.h>

// ---------------- problem constants ----------------
#define BB    32
#define HH    56
#define WW    56
#define DIM   192
#define HEADS 6
#define WS    7
#define KD    32
#define LL    (HH*WW)          // 3136
#define HIDDEN (4*DIM)         // 768
#define NWIN  64               // (56/7)^2
#define NTOK  49
#define HQKV  576
#define NWINS (BB*NWIN)        // 2048
#define SCALE 0.17677669529663689f

typedef __attribute__((ext_vector_type(16))) _Float16 v16h;
typedef __attribute__((ext_vector_type(8)))  _Float16 v8h;
typedef __attribute__((ext_vector_type(8)))  float    v8f;

// ---------------- WMMA helpers (CDNA5 gfx1250, wave32) ----------------
__device__ __forceinline__ v8f wmma_f16(v16h a, v16h b, v8f c) {
  return __builtin_amdgcn_wmma_f32_16x16x32_f16(false, a, false, b, (short)0, c, false, false);
}

// A-matrix 16x32 f16 fragment from row-major LDS buffer (stride S halves).
// lane: row m0+(lane&15); halves 0..15 = K kb..kb+15, kb = k0 + (lane>=16)*16
__device__ __forceinline__ v16h load_a_lds(const _Float16* buf, int S, int m0, int k0, int lane) {
  int row = m0 + (lane & 15);
  int kb  = k0 + ((lane >> 4) << 4);
  const _Float16* p = buf + row * S + kb;
  v16h r;
  *((v8h*)&r)       = *(const v8h*)(p);
  *(((v8h*)&r) + 1) = *(const v8h*)(p + 8);
  return r;
}

// B-matrix 32x16 f16 fragment where B[kk][n] = src[n0+n][koff+kk] (src row-major, stride S).
// lane: col = n0+(lane&15); halves j = kk = (lane>>4)*16 + j (contiguous in src row)
__device__ __forceinline__ v16h load_bt_lds(const _Float16* buf, int S, int koff, int n0, int lane) {
  int col = n0 + (lane & 15);
  int kb  = koff + ((lane >> 4) << 4);
  const _Float16* p = buf + col * S + kb;
  v16h r;
  *((v8h*)&r)       = *(const v8h*)(p);
  *(((v8h*)&r) + 1) = *(const v8h*)(p + 8);
  return r;
}

// Packed weight fragment (produced by k_pack): 32 bytes contiguous per lane
__device__ __forceinline__ v16h load_b_packed(const _Float16* w, int KT, int kt, int nt, int lane) {
  const _Float16* p = w + (((size_t)(nt * KT + kt)) * 32 + lane) * 16;
  v16h r;
  *((v8h*)&r)       = *(const v8h*)(p);
  *(((v8h*)&r) + 1) = *(const v8h*)(p + 8);
  return r;
}

// ---------------- K0: pack f32 weight [K][N] into WMMA B-fragment order, f16 ----------------
__global__ void k_pack(const float* __restrict__ w, _Float16* __restrict__ dst, int K, int N) {
  int idx = blockIdx.x * 256 + threadIdx.x;
  int total = K * N;
  if (idx >= total) return;
  int KT = K >> 5;
  int h    = idx & 15;
  int lane = (idx >> 4) & 31;
  int f    = idx >> 9;          // nt*KT + kt
  int kt   = f % KT;
  int nt   = f / KT;
  int kk   = kt * 32 + ((lane >> 4) << 4) + h;
  int c    = nt * 16 + (lane & 15);
  dst[idx] = (_Float16)w[(size_t)kk * N + c];
}

// ---------------- K1: fused LN1 + window attention (per-window block, 8 waves) ----------------
__global__ void __launch_bounds__(256)
k_attn(const float* __restrict__ x, const float* __restrict__ ln1_g, const float* __restrict__ ln1_b,
       const _Float16* __restrict__ wq, const float* __restrict__ qkv_b,
       const float* __restrict__ attn_bias, _Float16* __restrict__ obuf) {
  __shared__ _Float16 h16[64 * 200];   // LN1 output, padded rows 49..63 = 0   (25.6 KB)
  __shared__ _Float16 qkvh[64 * 72];   // q|k (cols 0..63) for current head    (9 KB)
  __shared__ _Float16 vt[32 * 72];     // v transposed: vt[d][token]           (4.5 KB)
  __shared__ _Float16 a16[64 * 72];    // scores -> softmax -> o staging       (9 KB)

  const int tid  = threadIdx.x;
  const int lane = tid & 31;
  const int wave = tid >> 5;

  const int w   = blockIdx.x;
  const int bat = w >> 6;
  const int win = w & 63;
  const int wi  = win >> 3;
  const int wj  = win & 7;

  // exercise global_prefetch_b8 on the packed QKV weights
  __builtin_prefetch(wq + (size_t)tid * 64, 0, 1);

  // ---- Stage A: LN1, one wave per row, 8 rows per wave ----
  for (int i = 0; i < 8; ++i) {
    int r = wave * 8 + i;
    if (r < NTOK) {
      int ti = r / WS, tj = r % WS;
      int l  = (wi * WS + ti) * WW + wj * WS + tj;
      const float* xr = x + ((size_t)bat * LL + l) * DIM;
      float v[6], s = 0.f, sq = 0.f;
#pragma unroll
      for (int j = 0; j < 6; ++j) {
        int c = lane + 32 * j;
        v[j] = xr[c];
        s += v[j]; sq += v[j] * v[j];
      }
#pragma unroll
      for (int off = 16; off > 0; off >>= 1) {
        s  += __shfl_xor(s, off, 32);
        sq += __shfl_xor(sq, off, 32);
      }
      float mean = s * (1.0f / DIM);
      float var  = sq * (1.0f / DIM) - mean * mean;
      float inv  = rsqrtf(var + 1e-5f);
#pragma unroll
      for (int j = 0; j < 6; ++j) {
        int c = lane + 32 * j;
        h16[r * 200 + c] = (_Float16)((v[j] - mean) * inv * ln1_g[c] + ln1_b[c]);
      }
    } else {
      for (int c = lane; c < 192; c += 32) h16[r * 200 + c] = (_Float16)0.f;
    }
  }
  __syncthreads();

  // ---- per-head loop ----
  for (int hd = 0; hd < HEADS; ++hd) {
    // (1) QKV GEMM for this head: M=64, N=96 (6 n-tiles), K=192. 24 tiles / 8 waves.
    //     n-tiles 0..3 -> q|k row-major in qkvh; n-tiles 4..5 -> v transposed into vt.
    for (int i = 0; i < 3; ++i) {
      int tile = wave + 8 * i;
      int mt = tile / 6, ntl = tile % 6;
      v8f acc = {};
#pragma unroll
      for (int kt = 0; kt < 6; ++kt) {
        v16h af = load_a_lds(h16, 200, mt * 16, kt * 32, lane);
        v16h bf = load_b_packed(wq, 6, kt, hd * 6 + ntl, lane);
        acc = wmma_f16(af, bf, acc);
      }
      int cl = ntl * 16 + (lane & 15);
      float bias = qkv_b[hd * 96 + cl];
      int rb = mt * 16 + ((lane >> 4) << 3);
      if (ntl < 4) {
#pragma unroll
        for (int j = 0; j < 8; ++j)
          qkvh[(rb + j) * 72 + cl] = (_Float16)(acc[j] + bias);
      } else {
        int d = cl - 64;                     // 0..31
        v8h pk;
#pragma unroll
        for (int j = 0; j < 8; ++j) pk[j] = (_Float16)(acc[j] + bias);
        *(v8h*)(vt + d * 72 + rb) = pk;      // one ds_store_b128 per lane
      }
    }
    __syncthreads();

    // (2) scores = q @ k^T : M=64, N=64, K=32 (single step). 16 tiles / 8 waves.
    for (int i = 0; i < 2; ++i) {
      int tile = wave * 2 + i;
      int mt = tile >> 2, nt = tile & 3;
      v8f acc = {};
      v16h af = load_a_lds(qkvh, 72, mt * 16, 0, lane);      // q at col offset 0
      v16h bf = load_bt_lds(qkvh, 72, KD, nt * 16, lane);    // k at col offset 32
      acc = wmma_f16(af, bf, acc);
      int rb = mt * 16 + ((lane >> 4) << 3);
      int col = nt * 16 + (lane & 15);
#pragma unroll
      for (int j = 0; j < 8; ++j) a16[(rb + j) * 72 + col] = (_Float16)acc[j];
    }
    __syncthreads();

    // (3) softmax in-place over a16 rows, with relative-position bias (idx = |di|*7+|dj|).
    if (tid < 64) {
      int n = tid;
      if (n < NTOK) {
        int ni = n / WS, nj = n % WS;
        float mx = -1e30f;
        float sc[NTOK];
        for (int m = 0; m < NTOK; ++m) {
          int mi = m / WS, mj = m % WS;
          int di = ni > mi ? ni - mi : mi - ni;
          int dj = nj > mj ? nj - mj : mj - nj;
          float s = (float)a16[n * 72 + m] * SCALE + attn_bias[hd * 49 + di * 7 + dj];
          sc[m] = s;
          mx = s > mx ? s : mx;
        }
        float den = 0.f;
        for (int m = 0; m < NTOK; ++m) { sc[m] = __expf(sc[m] - mx); den += sc[m]; }
        float rden = 1.0f / den;
        for (int m = 0; m < NTOK; ++m) a16[n * 72 + m] = (_Float16)(sc[m] * rden);
        for (int m = NTOK; m < 64; ++m) a16[n * 72 + m] = (_Float16)0.f;
      } else {
        for (int m = 0; m < 64; ++m) a16[n * 72 + m] = (_Float16)0.f;
      }
    }
    __syncthreads();

    // (4) o = a @ v : M=64, N=32 (2 n-tiles), K=64 (2 steps). 8 tiles / 8 waves.
    int mt = wave >> 1, nt2 = wave & 1;
    v8f oacc = {};
#pragma unroll
    for (int ks = 0; ks < 2; ++ks) {
      v16h af = load_a_lds(a16, 72, mt * 16, ks * 32, lane);
      v16h bf = load_bt_lds(vt, 72, ks * 32, nt2 * 16, lane);  // B[kk][d] = vt[d][kk], contiguous
      oacc = wmma_f16(af, bf, oacc);
    }
    __syncthreads();   // done reading a16; reuse it to stage o

    // (5) stage o tile [64][32] into a16's storage
    {
      _Float16* ost = a16;
      int rb = mt * 16 + ((lane >> 4) << 3);
#pragma unroll
      for (int j = 0; j < 8; ++j)
        ost[(rb + j) * 32 + nt2 * 16 + (lane & 15)] = (_Float16)oacc[j];
    }
    __syncthreads();

    // (6) coalesced writeback of this head's o columns (16B per thread)
    {
      const _Float16* ost = a16;
      int row = tid >> 2, ch = tid & 3;
      _Float16* op = obuf + (size_t)w * 64 * DIM + (size_t)row * DIM + hd * KD + ch * 8;
      *(v8h*)op = *(const v8h*)(ost + row * 32 + ch * 8);
    }
    __syncthreads();   // protect qkvh/vt/a16 before next head rewrites them
  }
}

// ---------------- K1b: per-window proj GEMM + bias + residual -> xa (f32) ----------------
__global__ void __launch_bounds__(256)
k_proj(const _Float16* __restrict__ obuf, const _Float16* __restrict__ wp,
       const float* __restrict__ proj_b, const float* __restrict__ x, float* __restrict__ xa) {
  __shared__ _Float16 o16[64 * 200];   // 25.6 KB
  const int tid  = threadIdx.x;
  const int lane = tid & 31;
  const int wave = tid >> 5;
  const int w    = blockIdx.x;
  const int bat  = w >> 6;
  const int win  = w & 63;
  const int wi   = win >> 3;
  const int wj   = win & 7;

  // stage head outputs into LDS via async copy: 64 rows x 24 x 16B chunks, 6 per thread
  const _Float16* src = obuf + (size_t)w * 64 * DIM;
  for (int idx = tid; idx < 64 * 24; idx += 256) {
    int row = idx / 24, ch = idx % 24;
    unsigned laddr = (unsigned)(unsigned long long)(const void*)(o16 + row * 200 + ch * 8);
    unsigned long long ga = (unsigned long long)(const void*)(src + (size_t)row * DIM + ch * 8);
    asm volatile("global_load_async_to_lds_b128 %0, %1, off"
                 :: "v"(laddr), "v"(ga)
                 : "memory");
  }
  asm volatile("s_wait_asynccnt 0x0" ::: "memory");
  __syncthreads();

  // proj: M=64, N=192 (12 n-tiles), K=192 (6 steps). 48 tiles / 8 waves.
  for (int i = 0; i < 6; ++i) {
    int tile = wave * 6 + i;
    int mt = tile / 12, nt = tile % 12;
    v8f acc = {};
#pragma unroll
    for (int kt = 0; kt < 6; ++kt) {
      v16h af = load_a_lds(o16, 200, mt * 16, kt * 32, lane);
      v16h bf = load_b_packed(wp, 6, kt, nt, lane);
      acc = wmma_f16(af, bf, acc);
    }
    int col = nt * 16 + (lane & 15);
    float pb = proj_b[col];
    int rb = mt * 16 + ((lane >> 4) << 3);
#pragma unroll
    for (int j = 0; j < 8; ++j) {
      int t = rb + j;
      if (t < NTOK) {
        int ti = t / WS, tj = t % WS;
        int l  = (wi * WS + ti) * WW + wj * WS + tj;
        size_t gi = ((size_t)bat * LL + l) * DIM + col;
        xa[gi] = acc[j] + pb + x[gi];
      }
    }
  }
}

// ---------------- K2: depthwise 3x3 conv + BN (float4 over channels) ----------------
__global__ void __launch_bounds__(256)
k_conv(const float* __restrict__ xa, const float* __restrict__ cw,
       const float* __restrict__ bn_g, const float* __restrict__ bn_b,
       const float* __restrict__ bn_m, const float* __restrict__ bn_v,
       float* __restrict__ xc) {
  int idx = blockIdx.x * 256 + threadIdx.x;
  const int C4 = DIM / 4;                           // 48
  if (idx >= BB * LL * C4) return;
  int c4 = idx % C4;
  int p  = idx / C4;
  int xx = p % WW; p /= WW;
  int yy = p % HH;
  int bat = p / HH;
  int c0 = c4 * 4;

  const float4* xa4 = (const float4*)xa;
  float a0 = 0.f, a1 = 0.f, a2 = 0.f, a3 = 0.f;
#pragma unroll
  for (int ky = 0; ky < 3; ++ky) {
    int ny = yy + ky - 1;
    if (ny < 0 || ny >= HH) continue;
#pragma unroll
    for (int kx = 0; kx < 3; ++kx) {
      int nx = xx + kx - 1;
      if (nx < 0 || nx >= WW) continue;
      float4 vv = xa4[((size_t)bat * LL + ny * WW + nx) * C4 + c4];
      int t = ky * 3 + kx;
      a0 += vv.x * cw[(c0 + 0) * 9 + t];
      a1 += vv.y * cw[(c0 + 1) * 9 + t];
      a2 += vv.z * cw[(c0 + 2) * 9 + t];
      a3 += vv.w * cw[(c0 + 3) * 9 + t];
    }
  }
  float4 o;
  o.x = (a0 - bn_m[c0 + 0]) * rsqrtf(bn_v[c0 + 0] + 1e-5f) * bn_g[c0 + 0] + bn_b[c0 + 0];
  o.y = (a1 - bn_m[c0 + 1]) * rsqrtf(bn_v[c0 + 1] + 1e-5f) * bn_g[c0 + 1] + bn_b[c0 + 1];
  o.z = (a2 - bn_m[c0 + 2]) * rsqrtf(bn_v[c0 + 2] + 1e-5f) * bn_g[c0 + 2] + bn_b[c0 + 2];
  o.w = (a3 - bn_m[c0 + 3]) * rsqrtf(bn_v[c0 + 3] + 1e-5f) * bn_g[c0 + 3] + bn_b[c0 + 3];
  ((float4*)xc)[((size_t)bat * LL + yy * WW + xx) * C4 + c4] = o;
}

// ---------------- K3: fused LN2 + FC1 + GELU + FC2 + residual (in-place over xc) ----------------
__global__ void __launch_bounds__(256)
k_mlp(const float* __restrict__ xc, const float* __restrict__ ln2_g, const float* __restrict__ ln2_b,
      const _Float16* __restrict__ w1, const float* __restrict__ fc1_b,
      const _Float16* __restrict__ w2, const float* __restrict__ fc2_b,
      float* __restrict__ out) {
  __shared__ _Float16 h16[64 * 200];   // LN2 output (25.6 KB)
  __shared__ _Float16 c16[64 * 72];    // hidden chunk (9 KB)

  const int tid  = threadIdx.x;
  const int lane = tid & 31;
  const int wave = tid >> 5;
  const size_t row0 = (size_t)blockIdx.x * 64;

  // ---- LN2, one wave per row ----
  for (int i = 0; i < 8; ++i) {
    int r = wave * 8 + i;
    const float* xr = xc + (row0 + r) * DIM;
    float v[6], s = 0.f, sq = 0.f;
#pragma unroll
    for (int j = 0; j < 6; ++j) {
      int c = lane + 32 * j;
      v[j] = xr[c];
      s += v[j]; sq += v[j] * v[j];
    }
#pragma unroll
    for (int off = 16; off > 0; off >>= 1) {
      s  += __shfl_xor(s, off, 32);
      sq += __shfl_xor(sq, off, 32);
    }
    float mean = s * (1.0f / DIM);
    float var  = sq * (1.0f / DIM) - mean * mean;
    float inv  = rsqrtf(var + 1e-5f);
#pragma unroll
    for (int j = 0; j < 6; ++j) {
      int c = lane + 32 * j;
      h16[r * 200 + c] = (_Float16)((v[j] - mean) * inv * ln2_g[c] + ln2_b[c]);
    }
  }
  __syncthreads();

  // ---- FC2 accumulators: 48 output tiles / 8 waves = 6 tiles per wave ----
  v8f acc[6];
#pragma unroll
  for (int t = 0; t < 6; ++t) acc[t] = (v8f){};

  // ---- hidden in chunks of 64 (12 chunks) ----
  for (int ch = 0; ch < 12; ++ch) {
    // FC1 chunk: 16 tiles / 8 waves
    for (int i = 0; i < 2; ++i) {
      int tile = wave * 2 + i;
      int mt = tile >> 2, ntl = tile & 3;
      int nt1 = ch * 4 + ntl;
      v8f a = {};
#pragma unroll
      for (int kt = 0; kt < 6; ++kt) {
        v16h af = load_a_lds(h16, 200, mt * 16, kt * 32, lane);
        v16h bf = load_b_packed(w1, 6, kt, nt1, lane);
        a = wmma_f16(af, bf, a);
      }
      int cl = ntl * 16 + (lane & 15);
      float bias = fc1_b[nt1 * 16 + (lane & 15)];
      int rb = mt * 16 + ((lane >> 4) << 3);
#pragma unroll
      for (int j = 0; j < 8; ++j) {
        float vv = a[j] + bias;
        vv = 0.5f * vv * (1.0f + erff(vv * 0.70710678118f));   // exact GELU
        c16[(rb + j) * 72 + cl] = (_Float16)vv;
      }
    }
    __syncthreads();

    // FC2 partial over this chunk
    for (int t = 0; t < 6; ++t) {
      int tile = wave * 6 + t;
      int mt = tile / 12, nt = tile % 12;
#pragma unroll
      for (int ks = 0; ks < 2; ++ks) {
        v16h af = load_a_lds(c16, 72, mt * 16, ks * 32, lane);
        v16h bf = load_b_packed(w2, 24, ch * 2 + ks, nt, lane);
        acc[t] = wmma_f16(af, bf, acc[t]);
      }
    }
    __syncthreads();   // before c16 is rewritten
  }

  // ---- store: out = xc + fc2 + fc2_b ----
  for (int t = 0; t < 6; ++t) {
    int tile = wave * 6 + t;
    int mt = tile / 12, nt = tile % 12;
    int col = nt * 16 + (lane & 15);
    float bias = fc2_b[col];
    int rb = mt * 16 + ((lane >> 4) << 3);
#pragma unroll
    for (int j = 0; j < 8; ++j) {
      size_t gi = (row0 + rb + j) * DIM + col;
      out[gi] = xc[gi] + acc[t][j] + bias;
    }
  }
}

// ---------------- workspace layout (bytes) ----------------
#define OFF_WQ 0u
#define SZ_WQ  (192u*576u*2u)                       // 221184
#define OFF_WP (OFF_WQ + SZ_WQ)
#define SZ_WP  (192u*192u*2u)                       // 73728
#define OFF_W1 (OFF_WP + SZ_WP)
#define SZ_W1  (192u*768u*2u)                       // 294912
#define OFF_W2 (OFF_W1 + SZ_W1)
#define SZ_W2  (768u*192u*2u)                       // 294912
#define OFF_OB (OFF_W2 + SZ_W2)                     // 884736
#define SZ_OB  ((size_t)NWINS*64u*DIM*2u)           // ~50.3 MB
#define OFF_XA (OFF_OB + SZ_OB)
#define SZ_XA  ((size_t)BB*LL*DIM*4u)               // ~77 MB

extern "C" void kernel_launch(void* const* d_in, const int* in_sizes, int n_in,
                              void* d_out, int out_size, void* d_ws, size_t ws_size,
                              hipStream_t stream) {
  const float* x       = (const float*)d_in[0];
  const float* ln1_g   = (const float*)d_in[1];
  const float* ln1_b   = (const float*)d_in[2];
  const float* qkv_w   = (const float*)d_in[3];
  const float* qkv_b   = (const float*)d_in[4];
  const float* proj_w  = (const float*)d_in[5];
  const float* proj_b  = (const float*)d_in[6];
  const float* attn_b  = (const float*)d_in[7];
  const float* conv_w  = (const float*)d_in[8];
  const float* bn_g    = (const float*)d_in[9];
  const float* bn_b    = (const float*)d_in[10];
  const float* bn_m    = (const float*)d_in[11];
  const float* bn_v    = (const float*)d_in[12];
  const float* ln2_g   = (const float*)d_in[13];
  const float* ln2_b   = (const float*)d_in[14];
  const float* fc1_w   = (const float*)d_in[15];
  const float* fc1_b   = (const float*)d_in[16];
  const float* fc2_w   = (const float*)d_in[17];
  const float* fc2_b   = (const float*)d_in[18];

  char* ws = (char*)d_ws;
  _Float16* WQ = (_Float16*)(ws + OFF_WQ);
  _Float16* WP = (_Float16*)(ws + OFF_WP);
  _Float16* W1 = (_Float16*)(ws + OFF_W1);
  _Float16* W2 = (_Float16*)(ws + OFF_W2);
  _Float16* OB = (_Float16*)(ws + OFF_OB);
  float*    XA = (float*)(ws + OFF_XA);
  float*    XC = (float*)d_out;                 // conv output lives in d_out; k_mlp rewrites in place

  k_pack<<<(192*576 + 255) / 256, 256, 0, stream>>>(qkv_w, WQ, 192, 576);
  k_pack<<<(192*192 + 255) / 256, 256, 0, stream>>>(proj_w, WP, 192, 192);
  k_pack<<<(192*768 + 255) / 256, 256, 0, stream>>>(fc1_w, W1, 192, 768);
  k_pack<<<(768*192 + 255) / 256, 256, 0, stream>>>(fc2_w, W2, 768, 192);

  k_attn<<<NWINS, 256, 0, stream>>>(x, ln1_g, ln1_b, WQ, qkv_b, attn_b, OB);
  k_proj<<<NWINS, 256, 0, stream>>>(OB, WP, proj_b, x, XA);
  k_conv<<<(BB * LL * (DIM / 4) + 255) / 256, 256, 0, stream>>>(XA, conv_w, bn_g, bn_b, bn_m, bn_v, XC);
  k_mlp<<<(BB * LL) / 64, 256, 0, stream>>>(XC, ln2_g, ln2_b, W1, fc1_b, W2, fc2_b, XC);
}
